// MORSpyMasterSmall_51135880626461
// MI455X (gfx1250) — matrix operand order repaired and checked by reading, not compiled, not run on hardware
//
#include <hip/hip_runtime.h>

typedef __attribute__((ext_vector_type(16))) __bf16 v16bf;
typedef __attribute__((ext_vector_type(8)))  __bf16 v8bf;
typedef __attribute__((ext_vector_type(8)))  float  v8f;
typedef __attribute__((ext_vector_type(2)))  float  v2f;

#define NBATCH 256
#define DIM    768
#define NPOS   9
#define NNEG   8
#define NNEUT  7
#define KTOP   80
#define NVOC   100000
#define NBINS  1024
#define CANDMAX 2048
#define NT2    (NVOC/32)   // 3125 n-tile-pairs for sims GEMM

// MLP dims (K = input dim, N = output dim), padded: K to x32, N to x16
#define K1 2304
#define N1 1800
#define N1P 1824
#define N2 1250
#define N2P 1280
#define N3 900
#define N3P 928
#define N4 768

// ---------------- workspace layout (bytes) ----------------
#define OFF_WT1  ((size_t)0)                         // 1824*2304*2
#define OFF_WT2  (OFF_WT1 + (size_t)N1P*K1*2)        // 1280*1824*2
#define OFF_WT3  (OFF_WT2 + (size_t)N2P*N1P*2)
#define OFF_WT4  (OFF_WT3 + (size_t)N3P*N2P*2)
#define OFF_X0   (OFF_WT4 + (size_t)N4*N3P*2)
#define OFF_H1   (OFF_X0  + (size_t)NBATCH*K1*2)
#define OFF_H2   (OFF_H1  + (size_t)NBATCH*N1P*2)
#define OFF_H3   (OFF_H2  + (size_t)NBATCH*N2P*2)
#define OFF_H4   (OFF_H3  + (size_t)NBATCH*N3P*2)    // fp32 pre-norm
#define OFF_MO   (OFF_H4  + (size_t)NBATCH*DIM*4)    // fp32 normalized model_out
#define OFF_HIST (OFF_MO  + (size_t)NBATCH*DIM*4)    // u32 256*1024
#define OFF_THR  (OFF_HIST+ (size_t)NBATCH*NBINS*4)  // i32 256
#define OFF_CCNT (OFF_THR + (size_t)1024)
#define OFF_CVAL (OFF_CCNT+ (size_t)1024)            // f32 256*2048
#define OFF_CIDX (OFF_CVAL+ (size_t)NBATCH*CANDMAX*4)
#define OFF_TIDX (OFF_CIDX+ (size_t)NBATCH*CANDMAX*4) // i32 256*80
#define OFF_TREW (OFF_TIDX+ (size_t)NBATCH*KTOP*4)    // f32 256*80
#define WS_NEED  (OFF_TREW+ (size_t)NBATCH*KTOP*4)

// ---------------- small utility kernels ----------------
__global__ void zero_u32(unsigned* p, int n) {
    int t = blockIdx.x * 256 + threadIdx.x;
    if (t < n) p[t] = 0u;
}

// Transpose + pad + fp32->bf16 weight conversion: W (K x N) -> Wt (Npad x Kpad)
__global__ void conv_w(const float* __restrict__ W, __bf16* __restrict__ Wt,
                       int K, int N, int Kpad, int Npad) {
    int t = blockIdx.x * 256 + threadIdx.x;
    int total = Npad * Kpad;
    if (t >= total) return;
    int n = t / Kpad, k = t - n * Kpad;
    float v = (n < N && k < K) ? W[(size_t)k * N + n] : 0.0f;
    Wt[t] = (__bf16)v;
}

// ---------------- stage 1: pooling ----------------
// x[b] = concat(l2norm(mean(neg)), l2norm(mean(neut)), l2norm(mean(pos))) -> bf16
__global__ __launch_bounds__(256)
void pool_embed(const float* __restrict__ pos, const float* __restrict__ neg,
                const float* __restrict__ neut, __bf16* __restrict__ X0) {
    __shared__ float mneg[DIM], mneut[DIM], mpos[DIM];
    __shared__ float red[256];
    __shared__ float inv3[3];
    int b = blockIdx.x, tid = threadIdx.x;
    for (int d = tid; d < DIM; d += 256) {
        float s = 0.f;
        for (int j = 0; j < NNEG; ++j)  s += neg[((size_t)b * NNEG + j) * DIM + d];
        mneg[d] = s / NNEG;
        s = 0.f;
        for (int j = 0; j < NNEUT; ++j) s += neut[((size_t)b * NNEUT + j) * DIM + d];
        mneut[d] = s / NNEUT;
        s = 0.f;
        for (int j = 0; j < NPOS; ++j)  s += pos[((size_t)b * NPOS + j) * DIM + d];
        mpos[d] = s / NPOS;
    }
    __syncthreads();
    const float* grp[3] = { mneg, mneut, mpos };
    for (int g = 0; g < 3; ++g) {
        float l = 0.f;
        for (int d = tid; d < DIM; d += 256) { float v = grp[g][d]; l += v * v; }
        red[tid] = l; __syncthreads();
        for (int s = 128; s > 0; s >>= 1) {
            if (tid < s) red[tid] += red[tid + s];
            __syncthreads();
        }
        if (tid == 0) inv3[g] = 1.0f / fmaxf(sqrtf(red[0]), 1e-12f);
        __syncthreads();
    }
    for (int d = tid; d < DIM; d += 256) {
        X0[(size_t)b * K1 + d]            = (__bf16)(mneg[d]  * inv3[0]);
        X0[(size_t)b * K1 + DIM + d]      = (__bf16)(mneut[d] * inv3[1]);
        X0[(size_t)b * K1 + 2 * DIM + d]  = (__bf16)(mpos[d]  * inv3[2]);
    }
}

// ---------------- stage 2: MLP GEMM (bf16 WMMA 16x16x32) ----------------
// C[M=256 x Npad] = A[256 x Kpad] * Bt[Npad x Kpad]^T + bias ; one 16x16 tile/wave
__global__ __launch_bounds__(128)
void mlp_gemm(const __bf16* __restrict__ A, const __bf16* __restrict__ Bt,
              const float* __restrict__ bias, __bf16* __restrict__ outb,
              float* __restrict__ outf, int Kpad, int N, int Npad, int relu) {
    int lane = threadIdx.x & 31;
    int gw = blockIdx.x * 4 + (threadIdx.x >> 5);
    int ntiles = Npad >> 4;
    int mt = gw / ntiles, nt = gw - mt * ntiles;
    int m0 = mt << 4, n0 = nt << 4;
    int lr = lane & 15, hi = lane >> 4;

    const __bf16* ar = A  + (size_t)(m0 + lr) * Kpad;
    const __bf16* br = Bt + (size_t)(n0 + lr) * Kpad;
    v8f c = {};
#pragma unroll 2
    for (int k0 = 0; k0 < Kpad; k0 += 32) {
        v8bf alo = *(const v8bf*)(ar + k0 + 8 * hi);
        v8bf ahi = *(const v8bf*)(ar + k0 + 16 + 8 * hi);
        v16bf a = __builtin_shufflevector(alo, ahi, 0,1,2,3,4,5,6,7,8,9,10,11,12,13,14,15);
        v8bf blo = *(const v8bf*)(br + k0 + 8 * hi);
        v8bf bhi = *(const v8bf*)(br + k0 + 16 + 8 * hi);
        v16bf bb = __builtin_shufflevector(blo, bhi, 0,1,2,3,4,5,6,7,8,9,10,11,12,13,14,15);
        c = __builtin_amdgcn_wmma_f32_16x16x32_bf16(false, a, false, bb, (short)0, c, false, false);
    }
    int n = n0 + lr;
    float bn = (n < N) ? bias[n] : 0.0f;
    if (relu) {
#pragma unroll
        for (int v = 0; v < 8; ++v) {
            int m = m0 + v + 8 * hi;
            float val = c[v] + bn;
            outb[(size_t)m * Npad + n] = (n < N) ? (__bf16)fmaxf(val, 0.0f) : (__bf16)0.0f;
        }
    } else {
#pragma unroll
        for (int v = 0; v < 8; ++v) {
            int m = m0 + v + 8 * hi;
            outf[(size_t)m * Npad + n] = c[v] + bn;
        }
    }
}

// l2-normalize rows of H4 -> MO (ws) and model_out (d_out slot 0)
__global__ __launch_bounds__(128)
void row_l2norm(const float* __restrict__ H4, float* __restrict__ MO,
                float* __restrict__ out0) {
    __shared__ float red[128];
    __shared__ float sc;
    int row = blockIdx.x, tid = threadIdx.x;
    const float* r = H4 + (size_t)row * DIM;
    float l = 0.f;
    for (int d = tid; d < DIM; d += 128) { float v = r[d]; l += v * v; }
    red[tid] = l; __syncthreads();
    for (int s = 64; s > 0; s >>= 1) { if (tid < s) red[tid] += red[tid + s]; __syncthreads(); }
    if (tid == 0) sc = 1.0f / fmaxf(sqrtf(red[0]), 1e-12f);
    __syncthreads();
    for (int d = tid; d < DIM; d += 128) {
        float v = r[d] * sc;
        MO[(size_t)row * DIM + d]   = v;
        out0[(size_t)row * DIM + d] = v;
    }
}

// ---------------- stage 3: sims GEMM (fp32 WMMA 16x16x4), 2 passes ----------------
__device__ __forceinline__ int simbin(float v) {
    int b = (int)((v + 1.0f) * 512.0f);
    return b < 0 ? 0 : (b > NBINS - 1 ? NBINS - 1 : b);
}

template<int MODE>  // 0 = histogram, 1 = collect candidates
__global__ __launch_bounds__(256)
void sims_pass(const float* __restrict__ A, const float* __restrict__ V,
               unsigned* __restrict__ hist, const int* __restrict__ rowThr,
               int* __restrict__ candCnt, float* __restrict__ candVal,
               int* __restrict__ candIdx) {
    int lane = threadIdx.x & 31;
    int gw = blockIdx.x * 8 + (threadIdx.x >> 5);
    int mt = gw / NT2, nt = gw - mt * NT2;
    int m0 = mt << 4, n0 = nt << 5;
    int lr = lane & 15, hi = lane >> 4;

    const float* ar  = A + (size_t)(m0 + lr) * DIM;
    const float* b0r = V + (size_t)(n0 + lr) * DIM;
    const float* b1r = V + (size_t)(n0 + 16 + lr) * DIM;
    v8f c0 = {}, c1 = {};
#pragma unroll 4
    for (int k = 0; k < DIM; k += 4) {
        // keep the HBM stream ahead of the MACs (lowers to global_prefetch_b8;
        // speculative TH -> safe even past the final vocab row)
        __builtin_prefetch(b0r + k + 64, 0, 1);
        __builtin_prefetch(b1r + k + 64, 0, 1);
        v2f a  = *(const v2f*)(ar  + k + 2 * hi);
        v2f b0 = *(const v2f*)(b0r + k + 2 * hi);
        v2f b1 = *(const v2f*)(b1r + k + 2 * hi);
        c0 = __builtin_amdgcn_wmma_f32_16x16x4_f32(false, a, false, b0, (short)0, c0, false, false);
        c1 = __builtin_amdgcn_wmma_f32_16x16x4_f32(false, a, false, b1, (short)0, c1, false, false);
    }
#pragma unroll
    for (int v = 0; v < 8; ++v) {
        int m = m0 + v + 8 * hi;
        float v0 = c0[v], v1 = c1[v];
        if (MODE == 0) {
            atomicAdd(&hist[(size_t)m * NBINS + simbin(v0)], 1u);
            atomicAdd(&hist[(size_t)m * NBINS + simbin(v1)], 1u);
        } else {
            int thr = rowThr[m];
            if (simbin(v0) >= thr) {
                int p = atomicAdd(&candCnt[m], 1);
                if (p < CANDMAX) { candVal[(size_t)m * CANDMAX + p] = v0; candIdx[(size_t)m * CANDMAX + p] = n0 + lr; }
            }
            if (simbin(v1) >= thr) {
                int p = atomicAdd(&candCnt[m], 1);
                if (p < CANDMAX) { candVal[(size_t)m * CANDMAX + p] = v1; candIdx[(size_t)m * CANDMAX + p] = n0 + 16 + lr; }
            }
        }
    }
}

__global__ void thr_scan(const unsigned* __restrict__ hist, int* __restrict__ rowThr) {
    int row = threadIdx.x;  // 256 threads, 1 block
    if (row >= NBATCH) return;
    int cum = 0, thr = 0;
    for (int b = NBINS - 1; b >= 0; --b) {
        cum += (int)hist[(size_t)row * NBINS + b];
        if (cum >= KTOP) { thr = b; break; }
    }
    rowThr[row] = thr;
}

// exact per-row top-80 (value desc, index asc) from <=2048 candidates
__global__ __launch_bounds__(256)
void topk_select(const int* __restrict__ candCnt, const float* __restrict__ candVal,
                 const int* __restrict__ candIdx, int* __restrict__ topIdx) {
    __shared__ float cv[CANDMAX];
    __shared__ int   ci[CANDMAX];
    __shared__ float rv[256];
    __shared__ int   ri[256], rp[256];
    int row = blockIdx.x, tid = threadIdx.x;
    int cnt = candCnt[row]; if (cnt > CANDMAX) cnt = CANDMAX;
    for (int t = tid; t < CANDMAX; t += 256) {
        if (t < cnt) { cv[t] = candVal[(size_t)row * CANDMAX + t]; ci[t] = candIdx[(size_t)row * CANDMAX + t]; }
        else         { cv[t] = -3.4e38f; ci[t] = 0x7fffffff; }
    }
    __syncthreads();
    for (int i = 0; i < KTOP; ++i) {
        float bv = -3.4e38f; int bi = 0x7fffffff, bp = -1;
        for (int t = tid; t < CANDMAX; t += 256) {
            float v = cv[t];
            if (v > bv || (v == bv && ci[t] < bi)) { bv = v; bi = ci[t]; bp = t; }
        }
        rv[tid] = bv; ri[tid] = bi; rp[tid] = bp;
        __syncthreads();
        for (int s = 128; s > 0; s >>= 1) {
            if (tid < s) {
                float ov = rv[tid + s]; int oi = ri[tid + s];
                if (ov > rv[tid] || (ov == rv[tid] && oi < ri[tid])) {
                    rv[tid] = ov; ri[tid] = oi; rp[tid] = rp[tid + s];
                }
            }
            __syncthreads();
        }
        if (tid == 0) {
            int wi = (ri[0] >= 0 && ri[0] < NVOC) ? ri[0] : 0;
            topIdx[(size_t)row * KTOP + i] = wi;
            if (rp[0] >= 0) cv[rp[0]] = -3.4e38f;
        }
        __syncthreads();
    }
}

// ---------------- stage 4: cosine scoring + reward ----------------
__global__ __launch_bounds__(64)
void cos_reward(const int* __restrict__ topIdx, const float* __restrict__ vocab,
                const float* __restrict__ pos, const float* __restrict__ neg,
                const float* __restrict__ neut, float* __restrict__ totRew) {
    __shared__ float we[DIM];
    __shared__ float redA[64], redB[64];
    __shared__ float cosj[24];
    __shared__ float wn2;
    int b = blockIdx.x / KTOP, k = blockIdx.x % KTOP;
    int tid = threadIdx.x;
    int wi = topIdx[(size_t)b * KTOP + k];
    if ((unsigned)wi >= (unsigned)NVOC) wi = 0;
    const float* wr = vocab + (size_t)wi * DIM;
    for (int d = tid; d < DIM; d += 64) we[d] = wr[d];
    __syncthreads();
    float l = 0.f;
    for (int d = tid; d < DIM; d += 64) l += we[d] * we[d];
    redA[tid] = l; __syncthreads();
    for (int s = 32; s > 0; s >>= 1) { if (tid < s) redA[tid] += redA[tid + s]; __syncthreads(); }
    if (tid == 0) wn2 = redA[0];
    __syncthreads();
    for (int j = 0; j < 24; ++j) {
        const float* e;
        if (j < NPOS)             e = pos  + ((size_t)b * NPOS  + j) * DIM;
        else if (j < NPOS + NNEG) e = neg  + ((size_t)b * NNEG  + (j - NPOS)) * DIM;
        else                      e = neut + ((size_t)b * NNEUT + (j - NPOS - NNEG)) * DIM;
        float dl = 0.f, sl = 0.f;
        for (int d = tid; d < DIM; d += 64) { float ev = e[d]; dl += ev * we[d]; sl += ev * ev; }
        redA[tid] = dl; redB[tid] = sl; __syncthreads();
        for (int s = 32; s > 0; s >>= 1) {
            if (tid < s) { redA[tid] += redA[tid + s]; redB[tid] += redB[tid + s]; }
            __syncthreads();
        }
        if (tid == 0) cosj[j] = redA[0] / fmaxf(sqrtf(wn2) * sqrtf(redB[0]), 1e-8f);
        __syncthreads();
    }
    if (tid == 0) {
        float maxNN = -3.4e38f;
        for (int j = NPOS; j < 24; ++j) maxNN = fmaxf(maxNN, cosj[j]);
        int primary = 0;
        for (int j = 0; j < NPOS; ++j) if (cosj[j] > maxNN) primary++;
        float maxNeg = -3.4e38f, maxNeut = -3.4e38f;
        for (int j = NPOS; j < NPOS + NNEG; ++j) maxNeg = fmaxf(maxNeg, cosj[j]);
        for (int j = NPOS + NNEG; j < 24; ++j) maxNeut = fmaxf(maxNeut, cosj[j]);
        float secondary = (maxNeg >= maxNeut) ? 0.0f : 1.0f;  // NEG_W=0, NEUT_W=1, tie->neg
        totRew[(size_t)b * KTOP + k] = (float)primary + secondary;
    }
}

// reward ranking + pooled outputs
__global__ __launch_bounds__(128)
void final_pool(const float* __restrict__ totRew, const int* __restrict__ topIdx,
                const float* __restrict__ vocab, float* __restrict__ outSearch,
                float* __restrict__ outMax, float* __restrict__ outMin) {
    __shared__ float tv[KTOP];
    __shared__ int wM[40], wA[40];
    __shared__ float meanM[DIM], meanA[DIM];
    __shared__ float redA[128], redB[128];
    __shared__ float scM, scA;
    __shared__ int bestw;
    int row = blockIdx.x, tid = threadIdx.x;
    if (tid < KTOP) tv[tid] = totRew[(size_t)row * KTOP + tid];
    __syncthreads();
    if (tid == 0) {
        bool used[KTOP];
        for (int j = 0; j < KTOP; ++j) used[j] = false;
        for (int i = 0; i < 40; ++i) {   // descending, tie -> lower index
            int best = -1;
            for (int j = 0; j < KTOP; ++j)
                if (!used[j] && (best < 0 || tv[j] > tv[best])) best = j;
            used[best] = true;
            int w = topIdx[(size_t)row * KTOP + best];
            if ((unsigned)w >= (unsigned)NVOC) w = 0;
            wM[i] = w;
            if (i == 0) bestw = w;
        }
        for (int j = 0; j < KTOP; ++j) used[j] = false;
        for (int i = 0; i < 40; ++i) {   // ascending, tie -> lower index
            int best = -1;
            for (int j = 0; j < KTOP; ++j)
                if (!used[j] && (best < 0 || tv[j] < tv[best])) best = j;
            used[best] = true;
            int w = topIdx[(size_t)row * KTOP + best];
            if ((unsigned)w >= (unsigned)NVOC) w = 0;
            wA[i] = w;
        }
    }
    __syncthreads();
    float lm = 0.f, la = 0.f;
    for (int d = tid; d < DIM; d += 128) {
        float sM = 0.f, sA = 0.f;
        for (int i = 0; i < 40; ++i) sM += vocab[(size_t)wM[i] * DIM + d];
        for (int i = 0; i < 40; ++i) sA += vocab[(size_t)wA[i] * DIM + d];
        sM *= (1.0f / 40.0f); sA *= (1.0f / 40.0f);
        meanM[d] = sM; meanA[d] = sA;
        lm += sM * sM; la += sA * sA;
    }
    redA[tid] = lm; redB[tid] = la; __syncthreads();
    for (int s = 64; s > 0; s >>= 1) {
        if (tid < s) { redA[tid] += redA[tid + s]; redB[tid] += redB[tid + s]; }
        __syncthreads();
    }
    if (tid == 0) {
        scM = 1.0f / fmaxf(sqrtf(redA[0]), 1e-12f);
        scA = 1.0f / fmaxf(sqrtf(redB[0]), 1e-12f);
    }
    __syncthreads();
    for (int d = tid; d < DIM; d += 128) {
        outMax[(size_t)row * DIM + d]    = meanM[d] * scM;
        outMin[(size_t)row * DIM + d]    = meanA[d] * scA;
        outSearch[(size_t)row * DIM + d] = vocab[(size_t)bestw * DIM + d];
    }
}

// ---------------- host ----------------
extern "C" void kernel_launch(void* const* d_in, const int* in_sizes, int n_in,
                              void* d_out, int out_size, void* d_ws, size_t ws_size,
                              hipStream_t stream) {
    (void)in_sizes; (void)n_in; (void)out_size;
    if (ws_size < WS_NEED) return;

    const float* pos   = (const float*)d_in[0];
    const float* neg   = (const float*)d_in[1];
    const float* neut  = (const float*)d_in[2];
    const float* W1 = (const float*)d_in[3];  const float* b1 = (const float*)d_in[4];
    const float* W2 = (const float*)d_in[5];  const float* b2 = (const float*)d_in[6];
    const float* W3 = (const float*)d_in[7];  const float* b3 = (const float*)d_in[8];
    const float* W4 = (const float*)d_in[9];  const float* b4 = (const float*)d_in[10];
    const float* vocab = (const float*)d_in[11];

    char* ws = (char*)d_ws;
    __bf16* WT1 = (__bf16*)(ws + OFF_WT1);
    __bf16* WT2 = (__bf16*)(ws + OFF_WT2);
    __bf16* WT3 = (__bf16*)(ws + OFF_WT3);
    __bf16* WT4 = (__bf16*)(ws + OFF_WT4);
    __bf16* X0  = (__bf16*)(ws + OFF_X0);
    __bf16* H1  = (__bf16*)(ws + OFF_H1);
    __bf16* H2  = (__bf16*)(ws + OFF_H2);
    __bf16* H3  = (__bf16*)(ws + OFF_H3);
    float*  H4  = (float*)(ws + OFF_H4);
    float*  MO  = (float*)(ws + OFF_MO);
    unsigned* HIST = (unsigned*)(ws + OFF_HIST);
    int*   THR  = (int*)(ws + OFF_THR);
    int*   CCNT = (int*)(ws + OFF_CCNT);
    float* CVAL = (float*)(ws + OFF_CVAL);
    int*   CIDX = (int*)(ws + OFF_CIDX);
    int*   TIDX = (int*)(ws + OFF_TIDX);
    float* TREW = (float*)(ws + OFF_TREW);

    float* out0 = (float*)d_out;                       // model_out
    float* out1 = out0 + (size_t)NBATCH * DIM;         // search_out
    float* out2 = out1 + (size_t)NBATCH * DIM;         // search_out_max
    float* out3 = out2 + (size_t)NBATCH * DIM;         // search_out_min

    // weight conversion (transpose + pad + bf16)
    conv_w<<<(N1P * K1 + 255) / 256, 256, 0, stream>>>(W1, WT1, K1, N1, K1, N1P);
    conv_w<<<(N2P * N1P + 255) / 256, 256, 0, stream>>>(W2, WT2, N1, N2, N1P, N2P);
    conv_w<<<(N3P * N2P + 255) / 256, 256, 0, stream>>>(W3, WT3, N2, N3, N2P, N3P);
    conv_w<<<(N4 * N3P + 255) / 256, 256, 0, stream>>>(W4, WT4, N3, N4, N3P, N4);

    // stage 1: pool
    pool_embed<<<NBATCH, 256, 0, stream>>>(pos, neg, neut, X0);

    // stage 2: MLP (bf16 WMMA)
    mlp_gemm<<<(16 * (N1P / 16)) / 4, 128, 0, stream>>>(X0, WT1, b1, H1, nullptr, K1,  N1, N1P, 1);
    mlp_gemm<<<(16 * (N2P / 16)) / 4, 128, 0, stream>>>(H1, WT2, b2, H2, nullptr, N1P, N2, N2P, 1);
    mlp_gemm<<<(16 * (N3P / 16)) / 4, 128, 0, stream>>>(H2, WT3, b3, H3, nullptr, N2P, N3, N3P, 1);
    mlp_gemm<<<(16 * (N4  / 16)) / 4, 128, 0, stream>>>(H3, WT4, b4, nullptr, H4, N3P, N4, N4, 0);
    row_l2norm<<<NBATCH, 128, 0, stream>>>(H4, MO, out0);

    // stage 3: sims (fp32 WMMA) + exact top-80 via histogram threshold
    zero_u32<<<(NBATCH * NBINS + 255) / 256, 256, 0, stream>>>(HIST, NBATCH * NBINS);
    zero_u32<<<1, 256, 0, stream>>>((unsigned*)CCNT, NBATCH);
    sims_pass<0><<<(16 * NT2) / 8, 256, 0, stream>>>(MO, vocab, HIST, THR, CCNT, CVAL, CIDX);
    thr_scan<<<1, 256, 0, stream>>>(HIST, THR);
    sims_pass<1><<<(16 * NT2) / 8, 256, 0, stream>>>(MO, vocab, HIST, THR, CCNT, CVAL, CIDX);
    topk_select<<<NBATCH, 256, 0, stream>>>(CCNT, CVAL, CIDX, TIDX);

    // stage 4: cosine rewards + final pooled outputs
    cos_reward<<<NBATCH * KTOP, 64, 0, stream>>>(TIDX, vocab, pos, neg, neut, TREW);
    final_pool<<<NBATCH, 128, 0, stream>>>(TREW, TIDX, vocab, out1, out2, out3);
}
